// GCN_10247791968964
// MI455X (gfx1250) — matrix operand order repaired and checked by reading, not compiled, not run on hardware
//
#include <hip/hip_runtime.h>

#define F_IN  128
#define F_HID 256
#define F_OUT 128

typedef __attribute__((ext_vector_type(16))) __bf16 v16bf;
typedef __attribute__((ext_vector_type(8)))  float  v8f;

// ---------------------------------------------------------------- utilities
__global__ void zero_kernel(float* __restrict__ p, long n) {
    long i = (long)blockIdx.x * blockDim.x + threadIdx.x;
    if (i < n) p[i] = 0.0f;
}

__global__ void degree_kernel(const int* __restrict__ dst,
                              float* __restrict__ deg, int E) {
    int e = blockIdx.x * blockDim.x + threadIdx.x;
    if (e < E) atomicAdd(&deg[dst[e]], 1.0f);
}

// One wave32 per edge: gather x[src] (L2-resident), scatter-add into agg[dst].
template<typename T, int F>
__global__ void scatter_kernel(const T* __restrict__ x,
                               const int* __restrict__ src,
                               const int* __restrict__ dst,
                               float* __restrict__ agg, int E) {
    int wave = blockIdx.x * (blockDim.x >> 5) + (threadIdx.x >> 5);
    int lane = threadIdx.x & 31;
    if (wave >= E) return;
    int s = src[wave];
    int d = dst[wave];
    const T* xp = x + (long)s * F;
    float*   ap = agg + (long)d * F;
#pragma unroll
    for (int f = lane; f < F; f += 32)
        atomicAdd(ap + f, (float)xp[f]);
}

// Pre-pack W[K,N] (row-major fp32) into WMMA B-fragment layout:
// Wp[((ntile*(K/32) + ks)*32 + lane)*16 + e] = bf16( W[(ks*32 + e + (lane>=16)*16) * N
//                                                     + ntile*16 + (lane&15)] )
// Each wave's B fragment is then 32 lanes x 16 contiguous bf16 (2x b128/lane).
template<int K, int N>
__global__ void pack_w_kernel(const float* __restrict__ W, __bf16* __restrict__ Wp) {
    int i = blockIdx.x * blockDim.x + threadIdx.x;
    if (i >= K * N) return;
    int e    = i & 15;
    int lane = (i >> 4) & 31;
    int ks   = (i >> 9) % (K / 32);
    int nt   = (i >> 9) / (K / 32);
    int k = ks * 32 + e + ((lane >= 16) ? 16 : 0);
    int n = nt * 16 + (lane & 15);
    Wp[i] = (__bf16)W[k * N + n];
}

// ------------------------------------------------- fused SAGE layer (WMMA)
// Out[16 nodes x N] = act( Aself@Wself + (agg/max(deg,1))@Wneigh + b )
// Block: 16 nodes, N/16 waves; each wave owns one 16-wide column tile.
template<int K, int N, bool SIG, typename TIN, typename TOUT>
__global__ void __launch_bounds__(32 * (N / 16))
sage_layer_kernel(const TIN*   __restrict__ Aself,   // [nodes,K]
                  const float* __restrict__ agg,     // [nodes,K]
                  const float* __restrict__ deg,     // [nodes]
                  const __bf16* __restrict__ WselfP, // packed B frags
                  const __bf16* __restrict__ WneighP,// packed B frags
                  const float* __restrict__ bias,    // [N]
                  TOUT* __restrict__ Out)            // [nodes,N]
{
    constexpr int NW  = N / 16;
    constexpr int KS  = K / 32;
    constexpr int LDA = K + 8;                 // pad: 16B -> shift 4 banks/row
    __shared__ __bf16 As[16 * LDA];
    __shared__ __bf16 An[16 * LDA];
    __shared__ float  invd[16];

    const int tid   = threadIdx.x;
    const int node0 = blockIdx.x * 16;

    if (tid < 16) {
        float d = deg[node0 + tid];
        invd[tid] = 1.0f / (d > 1.0f ? d : 1.0f);
    }
    __syncthreads();

    // Stage A tiles (self + mean-neighbor) in LDS as bf16.
    for (int i = tid; i < 16 * K; i += 32 * NW) {
        int r = i / K, c = i - r * K;
        long idx = (long)(node0 + r) * K + c;
        As[r * LDA + c] = (__bf16)(float)Aself[idx];
        An[r * LDA + c] = (__bf16)(agg[idx] * invd[r]);
    }
    __syncthreads();

    const int lane  = tid & 31;
    const int ntile = tid >> 5;
    const int ncol  = ntile * 16 + (lane & 15);
    const int m     = lane & 15;
    const int khiA  = (lane >= 16) ? 16 : 0;   // contiguous k = 16h + e (matches packed B)

    const v16bf* Bs = (const v16bf*)WselfP  + (long)ntile * KS * 32 + lane;
    const v16bf* Bn = (const v16bf*)WneighP + (long)ntile * KS * 32 + lane;

    v8f acc = {};
#pragma unroll
    for (int ks = 0; ks < KS; ++ks) {
        const int k0 = ks * 32;
        v16bf a_s, a_n;
#pragma unroll
        for (int e = 0; e < 16; ++e) {
            int ka = k0 + e + khiA;
            a_s[e] = As[m * LDA + ka];
            a_n[e] = An[m * LDA + ka];
        }
        v16bf b_s = Bs[ks * 32];               // 32B/lane, coalesced, L1-hot
        v16bf b_n = Bn[ks * 32];
        acc = __builtin_amdgcn_wmma_f32_16x16x32_bf16(
                  false, a_s, false, b_s, (short)0, acc, false, false);
        acc = __builtin_amdgcn_wmma_f32_16x16x32_bf16(
                  false, a_n, false, b_n, (short)0, acc, false, false);
    }

    const float bv    = bias[ncol];
    const int   mbase = (lane >= 16) ? 8 : 0;  // ISA 32-bit C/D layout
#pragma unroll
    for (int r = 0; r < 8; ++r) {
        float v = acc[r] + bv;
        if (SIG) v = 1.0f / (1.0f + __expf(-v));
        Out[(long)(node0 + mbase + r) * N + ncol] = (TOUT)v;
    }
}

// ---------------------------------------------------------------- launcher
extern "C" void kernel_launch(void* const* d_in, const int* in_sizes, int n_in,
                              void* d_out, int out_size, void* d_ws, size_t ws_size,
                              hipStream_t stream) {
    const float* x    = (const float*)d_in[0];
    const int*   esrc = (const int*)  d_in[1];
    const int*   edst = (const int*)  d_in[2];
    const float* Ws1  = (const float*)d_in[3];
    const float* Wn1  = (const float*)d_in[4];
    const float* b1   = (const float*)d_in[5];
    const float* Ws2  = (const float*)d_in[6];
    const float* Wn2  = (const float*)d_in[7];
    const float* b2   = (const float*)d_in[8];
    float*       out  = (float*)d_out;

    const int E  = in_sizes[1];
    const int Nn = in_sizes[0] / F_IN;        // 100000, divisible by 16

    // Workspace: deg | h(bf16) | agg(fp32, F_HID-sized, reused) | packed W x4
    char* w = (char*)d_ws;
    auto aln = [](size_t v) { return (v + 4095) & ~(size_t)4095; };
    size_t off = 0;
    float*  deg  = (float*)(w + off);  off = aln(off + (size_t)Nn * sizeof(float));
    __bf16* h    = (__bf16*)(w + off); off = aln(off + (size_t)Nn * F_HID * sizeof(__bf16));
    float*  agg  = (float*)(w + off);  off = aln(off + (size_t)Nn * F_HID * sizeof(float));
    __bf16* p1s  = (__bf16*)(w + off); off = aln(off + (size_t)F_IN  * F_HID * sizeof(__bf16));
    __bf16* p1n  = (__bf16*)(w + off); off = aln(off + (size_t)F_IN  * F_HID * sizeof(__bf16));
    __bf16* p2s  = (__bf16*)(w + off); off = aln(off + (size_t)F_HID * F_OUT * sizeof(__bf16));
    __bf16* p2n  = (__bf16*)(w + off);

    const long nAgg1 = (long)Nn * F_IN;
    const long nAgg2 = (long)Nn * F_HID;

    // ---- weight packing (tiny: 4 x <=32K elements) ----
    pack_w_kernel<F_IN,  F_HID><<<(F_IN  * F_HID + 255) / 256, 256, 0, stream>>>(Ws1, p1s);
    pack_w_kernel<F_IN,  F_HID><<<(F_IN  * F_HID + 255) / 256, 256, 0, stream>>>(Wn1, p1n);
    pack_w_kernel<F_HID, F_OUT><<<(F_HID * F_OUT + 255) / 256, 256, 0, stream>>>(Ws2, p2s);
    pack_w_kernel<F_HID, F_OUT><<<(F_HID * F_OUT + 255) / 256, 256, 0, stream>>>(Wn2, p2n);

    // ---- layer 1 ----
    zero_kernel<<<(Nn + 255) / 256, 256, 0, stream>>>(deg, (long)Nn);
    zero_kernel<<<(int)((nAgg1 + 255) / 256), 256, 0, stream>>>(agg, nAgg1);
    degree_kernel<<<(E + 255) / 256, 256, 0, stream>>>(edst, deg, E);
    scatter_kernel<float, F_IN><<<(E + 7) / 8, 256, 0, stream>>>(x, esrc, edst, agg, E);
    sage_layer_kernel<F_IN, F_HID, true, float, __bf16>
        <<<Nn / 16, 32 * (F_HID / 16), 0, stream>>>(x, agg, deg, p1s, p1n, b1, h);

    // ---- layer 2 ----
    zero_kernel<<<(int)((nAgg2 + 255) / 256), 256, 0, stream>>>(agg, nAgg2);
    scatter_kernel<__bf16, F_HID><<<(E + 7) / 8, 256, 0, stream>>>(h, esrc, edst, agg, E);
    sage_layer_kernel<F_HID, F_OUT, false, __bf16, float>
        <<<Nn / 16, 32 * (F_OUT / 16), 0, stream>>>(h, agg, deg, p2s, p2n, b2, out);
}
